// QuantDeformConv2d_70901320123223
// MI455X (gfx1250) — compile-verified, matrix operand order checked by reading, not run on hardware
//
#include <hip/hip_runtime.h>

typedef __attribute__((ext_vector_type(16))) _Float16 v16h;
typedef __attribute__((ext_vector_type(8)))  _Float16 v8h;
typedef __attribute__((ext_vector_type(2)))  _Float16 v2h;
typedef __attribute__((ext_vector_type(8)))  float    v8f;
typedef __attribute__((ext_vector_type(4)))  int      v4i;
typedef __attribute__((ext_vector_type(4)))  float    v4f;

#define N_      8
#define CIN_    64
#define H_      128
#define W_      128
#define COUT_   64
#define KTAP    9
#define HO_     128
#define WO_     128
#define KDIM    (KTAP * CIN_)   // 576
#define NCHUNK  (KDIM / 32)     // 18
#define TSTRIDE 40              // padded halves per pixel row (conflict-free b128)

__device__ __forceinline__ int iclamp(int v, int lo, int hi) {
    return v < lo ? lo : (v > hi ? hi : v);
}

// ---------------- Phase 1: quantize weight -> WMMA-A-layout f16 in ws ----------------
// A layout for v_wmma_f32_16x16x32_f16 (16-bit A 16x32):
//   lane l: M = l&15 ; halves j<8  -> K = (l>>4)*8 + j
//                      halves j>=8 -> K = 16 + (l>>4)*8 + (j-8)
// ws layout: [ct(4)][chunk(18)][lane(32)][j(16)] halves, K-dim order kk = tap*64 + cin.
__global__ void quant_weight_kernel(const float* __restrict__ w,
                                    _Float16* __restrict__ wsA) {
    __shared__ float red[256];
    __shared__ float s_scale;
    int tid = threadIdx.x;
    float m = 0.f;
    for (int i = tid; i < COUT_ * CIN_ * KTAP; i += 256) m = fmaxf(m, fabsf(w[i]));
    red[tid] = m;
    __syncthreads();
    for (int s = 128; s > 0; s >>= 1) {
        if (tid < s) red[tid] = fmaxf(red[tid], red[tid + s]);
        __syncthreads();
    }
    if (tid == 0) s_scale = fmaxf(red[0], 1e-8f) / 127.f;
    __syncthreads();
    float scale = s_scale;

    for (int e = tid; e < 4 * NCHUNK * 32 * 16; e += 256) {
        int j   = e & 15;
        int l   = (e >> 4) & 31;
        int c   = (e >> 9) % NCHUNK;
        int ct  = e / (NCHUNK * 512);
        int hi  = l >> 4;
        int kkl = (j < 8) ? (hi * 8 + j) : (16 + hi * 8 + (j - 8));
        int kk  = c * 32 + kkl;
        int tap = kk >> 6;
        int cin = kk & 63;
        int cout = ct * 16 + (l & 15);
        float wv = w[(cout * CIN_ + cin) * KTAP + tap];
        float q = rintf(wv / scale);           // round-to-nearest-even like jnp.round
        q = fminf(fmaxf(q, -128.f), 127.f);
        wsA[e] = (_Float16)(q * scale);
    }
}

// ---------------- Phase 2: deformable gather + WMMA GEMM ----------------
__global__ __launch_bounds__(128)
void deform_conv_wmma_kernel(const float* __restrict__ x,
                             const float* __restrict__ off,
                             const _Float16* __restrict__ wsA,
                             float* __restrict__ out) {
    // per-(tap,pixel) record: [i00,i01,i10,i11][w00,w01,w10,w11], 16B aligned
    __shared__ __align__(16) int   s_idx[KTAP * 64][4];
    __shared__ __align__(16) float s_wgt[KTAP * 64][4];
    __shared__ _Float16 tile[2][64 * TSTRIDE];

    const int tid = threadIdx.x;
    const int bid = blockIdx.x;
    const int n  = bid >> 8;          // 256 tiles per image
    const int tY = (bid >> 4) & 15;
    const int tX = bid & 15;
    const int ho0 = tY * 8, wo0 = tX * 8;

    // --- precompute bilinear corner linear indices + validity-masked weights ---
    for (int e = tid; e < KTAP * 64; e += 128) {
        int tap = e >> 6;
        int p   = e & 63;
        int ho = ho0 + (p >> 3);
        int wo = wo0 + (p & 7);
        int ky = tap / 3, kx = tap % 3;
        float oy = off[(((n * 2 * KTAP) + 2 * tap    ) * HO_ + ho) * WO_ + wo];
        float ox = off[(((n * 2 * KTAP) + 2 * tap + 1) * HO_ + ho) * WO_ + wo];
        float py = (float)(ho - 1 + ky) + oy;   // stride 1, pad 1, dil 1
        float px = (float)(wo - 1 + kx) + ox;
        float fy = floorf(py), fx = floorf(px);
        int y0 = (int)fy, x0 = (int)fx;
        float dy = py - fy, dx = px - fx;
        float vy0 = (y0 >= 0     && y0 < H_)     ? 1.f : 0.f;
        float vy1 = (y0 + 1 >= 0 && y0 + 1 < H_) ? 1.f : 0.f;
        float vx0 = (x0 >= 0     && x0 < W_)     ? 1.f : 0.f;
        float vx1 = (x0 + 1 >= 0 && x0 + 1 < W_) ? 1.f : 0.f;
        int ya = iclamp(y0, 0, H_ - 1),  yb = iclamp(y0 + 1, 0, H_ - 1);
        int xa = iclamp(x0, 0, W_ - 1),  xe = iclamp(x0 + 1, 0, W_ - 1);
        s_idx[e][0] = ya * W_ + xa;
        s_idx[e][1] = ya * W_ + xe;
        s_idx[e][2] = yb * W_ + xa;
        s_idx[e][3] = yb * W_ + xe;
        s_wgt[e][0] = (1.f - dy) * (1.f - dx) * vy0 * vx0;
        s_wgt[e][1] = (1.f - dy) * dx         * vy0 * vx1;
        s_wgt[e][2] = dy         * (1.f - dx) * vy1 * vx0;
        s_wgt[e][3] = dy         * dx         * vy1 * vx1;
    }
    __syncthreads();

    const float* xb = x + (size_t)n * CIN_ * H_ * W_;

    // fill one 32(K) x 64(px) f16 tile; chunk c: tap = c>>1, cin half = (c&1)*32
    auto fill = [&](int c, int buf) {
        int tap  = c >> 1;
        int cin0 = (c & 1) * 32;
        #pragma unroll
        for (int i = 0; i < 8; ++i) {
            int e   = i * 128 + tid;      // 0..1023 : (px, kk-pair)
            int p   = e & 63;
            int kk2 = (e >> 6) * 2;       // 0,2,..,30
            int ent = tap * 64 + p;
            v4i ci = *(const v4i*)s_idx[ent];   // ds_load_b128 (broadcast)
            v4f cw = *(const v4f*)s_wgt[ent];   // ds_load_b128 (broadcast)
            const float* xc = xb + (cin0 + kk2) * (H_ * W_);
            const float* xd = xc + H_ * W_;
            float va = cw.x * xc[ci.x] + cw.y * xc[ci.y]
                     + cw.z * xc[ci.z] + cw.w * xc[ci.w];
            float vb = cw.x * xd[ci.x] + cw.y * xd[ci.y]
                     + cw.z * xd[ci.z] + cw.w * xd[ci.w];
            v2h pk;
            pk.x = (_Float16)va;
            pk.y = (_Float16)vb;
            *(v2h*)&tile[buf][p * TSTRIDE + kk2] = pk;
        }
    };

    const int wave = tid >> 5;   // cout tile 0..3
    const int lane = tid & 31;

    v8f zero = {0.f, 0.f, 0.f, 0.f, 0.f, 0.f, 0.f, 0.f};
    v8f acc[4] = {zero, zero, zero, zero};

    fill(0, 0);
    __syncthreads();

    for (int c = 0; c < NCHUNK; ++c) {
        int buf = c & 1;
        if (c + 1 < NCHUNK) fill(c + 1, buf ^ 1);

        // A: contiguous 32B per lane, pre-swizzled in ws
        v16h a = *(const v16h*)(wsA + ((size_t)(wave * NCHUNK + c) * 32 + lane) * 16);

        #pragma unroll
        for (int pt = 0; pt < 4; ++pt) {
            // B 32x16 f16, mirrored layout: lane -> N = lane&15,
            // halves j<8 -> K=(lane>>4)*8+j ; j>=8 -> K=16+(lane>>4)*8+(j-8)
            const _Float16* bp =
                &tile[buf][(pt * 16 + (lane & 15)) * TSTRIDE + (lane >> 4) * 8];
            v8h blo = *(const v8h*)bp;          // ds_load_b128
            v8h bhi = *(const v8h*)(bp + 16);   // ds_load_b128
            v16h b = __builtin_shufflevector(blo, bhi,
                     0, 1, 2, 3, 4, 5, 6, 7, 8, 9, 10, 11, 12, 13, 14, 15);
            acc[pt] = __builtin_amdgcn_wmma_f32_16x16x32_f16(
                false, a, false, b, (short)0, acc[pt], false, false);
        }
        __syncthreads();
    }

    // --- writeback: D VGPR r, lane -> M = r + (lane>>4)*8, N = lane&15 ---
    const int m0 = (lane >> 4) * 8;
    const int nn = lane & 15;
    #pragma unroll
    for (int pt = 0; pt < 4; ++pt) {
        int p  = pt * 16 + nn;
        int ho = ho0 + (p >> 3);
        int wo = wo0 + (p & 7);
        float* op = out + (size_t)n * COUT_ * HO_ * WO_ + ho * WO_ + wo;
        #pragma unroll
        for (int r = 0; r < 8; ++r) {
            int cout = wave * 16 + m0 + r;
            op[(size_t)cout * HO_ * WO_] = acc[pt][r];
        }
    }
}

extern "C" void kernel_launch(void* const* d_in, const int* in_sizes, int n_in,
                              void* d_out, int out_size, void* d_ws, size_t ws_size,
                              hipStream_t stream) {
    const float* x      = (const float*)d_in[0];
    const float* offset = (const float*)d_in[1];
    const float* weight = (const float*)d_in[2];
    _Float16*    wsA    = (_Float16*)d_ws;    // 36864 halves = 73728 B
    float*       out    = (float*)d_out;

    quant_weight_kernel<<<1, 256, 0, stream>>>(weight, wsA);

    // 8 images * 16x16 tiles of 8x8 pixels = 2048 blocks, 128 threads (4 waves)
    deform_conv_wmma_kernel<<<2048, 128, 0, stream>>>(x, offset, wsA, out);
}
